// MultiLayerAttentionModel_75041668596016
// MI455X (gfx1250) — compile-verified, matrix operand order checked by reading, not run on hardware
//
#include <hip/hip_runtime.h>
#include <hip/hip_bf16.h>

// ---------------------------------------------------------------------------
// MI455X (gfx1250) implementation of the 4-layer linear-attention model.
// GEMMs: v_wmma_f32_16x16x32_bf16 (f32 accumulate). LDS tiles fed by
// global_load_async_to_lds_b128 (ASYNCcnt) or the Tensor Data Mover when
// the builtins exist; uint4 staging fallback otherwise.
// ---------------------------------------------------------------------------

typedef __bf16 bf16;
typedef float v8f   __attribute__((ext_vector_type(8)));
typedef bf16  v16bf __attribute__((ext_vector_type(16)));
typedef unsigned int u32x4 __attribute__((ext_vector_type(4)));
typedef int  i32x8 __attribute__((ext_vector_type(8)));
typedef int  i32x4 __attribute__((ext_vector_type(4)));

// LOAD_MODE: 1 = per-lane async global->LDS, 2 = TDM (ROCm7.2 5-arg), 0 = plain
#if __has_builtin(__builtin_amdgcn_global_load_async_to_lds_b128) && \
    __has_builtin(__builtin_amdgcn_s_wait_asynccnt)
#define LOAD_MODE 1
#elif defined(__clang_major__) && (__clang_major__ <= 22) && \
    __has_builtin(__builtin_amdgcn_tensor_load_to_lds) && \
    __has_builtin(__builtin_amdgcn_s_wait_tensorcnt)
#define LOAD_MODE 2
#else
#define LOAD_MODE 0
#endif

#if LOAD_MODE == 1
// Builtin takes pointers to 16B int vectors: AS(1) source, AS(3) LDS dest.
typedef int v4i_vs __attribute__((vector_size(16)));
typedef __attribute__((address_space(1))) v4i_vs gv4i_t;
typedef __attribute__((address_space(3))) v4i_vs lv4i_t;
#define AS_GLOBAL(p) ((gv4i_t*)(unsigned long long)(__SIZE_TYPE__)(p))
#define AS_LDS(p)    ((lv4i_t*)(unsigned)(__SIZE_TYPE__)(p))
#endif

// Problem constants
#define BB   64
#define NN   2047
#define SS   2048          // tokens per batch (N + 1 query)
#define DD   512
#define TT   513           // token dim
#define HH   512
#define KP   544           // TT padded to multiple of 32 (K-dim)
#define NP   576           // TT padded to multiple of 64 (N-dim)
#define LL   4
#define LN_EPS   1e-5f
#define ATTN_EPS 1e-6f

// ---------------------------------------------------------------------------
// Block-wide sum over 256 threads (8 wave32 waves)
// ---------------------------------------------------------------------------
__device__ __forceinline__ float blockReduceSum(float v, float* sm) {
#pragma unroll
  for (int o = 16; o > 0; o >>= 1) v += __shfl_down(v, o, 32);
  const int lane = threadIdx.x & 31, wid = threadIdx.x >> 5;
  if (lane == 0) sm[wid] = v;
  __syncthreads();
  if (wid == 0) {
    v = (lane < 8) ? sm[lane] : 0.f;
#pragma unroll
    for (int o = 4; o > 0; o >>= 1) v += __shfl_down(v, o, 32);
    if (lane == 0) sm[0] = v;
  }
  __syncthreads();
  v = sm[0];
  __syncthreads();
  return v;
}

// ---------------------------------------------------------------------------
// Build padded token tensor: tok[b][n][0..543] f32
// ---------------------------------------------------------------------------
__global__ __launch_bounds__(256) void build_tokens(
    const float* __restrict__ xs, const float* __restrict__ ys,
    const float* __restrict__ qx, float* __restrict__ tok) {
  const long long row = blockIdx.x;
  const int b = (int)(row >> 11), n = (int)(row & (SS - 1));
  float* o = tok + row * KP;
#pragma unroll
  for (int i = 0; i < 3; ++i) {
    const int t = threadIdx.x + i * 256;
    if (t >= KP) break;
    float v = 0.f;
    if (t < DD)
      v = (n < NN) ? xs[((long long)b * NN + n) * DD + t] : qx[(long long)b * DD + t];
    else if (t == DD && n < NN)
      v = ys[(long long)b * NN + n];
    o[t] = v;
  }
}

// ---------------------------------------------------------------------------
// Pad+convert f32 weight [ri][ci] -> bf16 [ro][co], zero pads
// ---------------------------------------------------------------------------
__global__ __launch_bounds__(256) void padconv(
    const float* __restrict__ in, int ri, int ci,
    bf16* __restrict__ outp, int ro, int co) {
  const int idx = blockIdx.x * 256 + threadIdx.x;
  if (idx >= ro * co) return;
  const int r = idx / co, c = idx % co;
  outp[idx] = (r < ri && c < ci) ? (bf16)in[r * ci + c] : (bf16)0.f;
}

// ---------------------------------------------------------------------------
// LayerNorm over first 513 cols of a 544-col f32 row -> bf16 row (zero pads)
// ---------------------------------------------------------------------------
__global__ __launch_bounds__(256) void ln_to_bf16(
    const float* __restrict__ tok, const float* __restrict__ w,
    const float* __restrict__ bb, bf16* __restrict__ outp) {
  __shared__ float sm[16];
  const long long row = blockIdx.x;
  const float* x = tok + row * KP;
  float v[3], s = 0.f, ss = 0.f;
#pragma unroll
  for (int i = 0; i < 3; ++i) {
    const int t = threadIdx.x + i * 256;
    const float val = (t < TT) ? x[t] : 0.f;
    v[i] = val; s += val; ss += val * val;
  }
  s = blockReduceSum(s, sm);
  ss = blockReduceSum(ss, sm);
  const float mean = s * (1.f / TT);
  const float var  = ss * (1.f / TT) - mean * mean;
  const float inv  = rsqrtf(var + LN_EPS);
  bf16* o = outp + row * KP;
#pragma unroll
  for (int i = 0; i < 3; ++i) {
    const int t = threadIdx.x + i * 256;
    if (t < KP) o[t] = (t < TT) ? (bf16)((v[i] - mean) * inv * w[t] + bb[t]) : (bf16)0.f;
  }
}

#if LOAD_MODE == 2
// ---------------------------------------------------------------------------
// TDM: DMA a [rows x 32] bf16 tile (row stride = strideElems) from global
// into LDS rows of 40 halves (64B data + 16B pad), per CDNA5 D# layout.
// ---------------------------------------------------------------------------
__device__ __forceinline__ void tdm_load_tile(
    const bf16* gaddr, int strideElems, int rows, unsigned ldsByteOff) {
  const unsigned long long ga = (unsigned long long)(__SIZE_TYPE__)gaddr;
  u32x4 g0;
  g0.x = 1u;                                        // count=1, user mode
  g0.y = ldsByteOff;                                // lds_addr
  g0.z = (unsigned)(ga & 0xffffffffu);              // global_addr[31:0]
  g0.w = (unsigned)((ga >> 32) & 0x01ffffffu) | (2u << 30);  // addr[56:32], type=2
  const unsigned td0 = 1u << 20, td1 = 1u << 20;    // huge tensor dims (no clip)
  i32x8 g1;
  g1[0] = (1 << 16)        // data_size = 2 bytes
        | (1 << 20)        // pad_enable
        | (3 << 22)        // pad_interval: 16 DWORDs (64B of data)
        | (3 << 25);       // pad_amount: 4 DWORDs (16B pad)
  g1[1] = (int)((td0 & 0xffffu) << 16);             // tensor_dim0[15:0]
  g1[2] = (int)(((td0 >> 16) & 0xffffu) | ((td1 & 0xffffu) << 16));
  g1[3] = (int)(((td1 >> 16) & 0xffffu) | (32u << 16));      // tile_dim0 = 32
  g1[4] = rows & 0xffff;                             // tile_dim1, tile_dim2=0
  g1[5] = (int)(unsigned)strideElems;                // tensor_dim0_stride[31:0]
  g1[6] = 0;
  g1[7] = 0;
  const i32x4 z4 = {0, 0, 0, 0};
  __builtin_amdgcn_tensor_load_to_lds(g0, g1, z4, z4, 0);
}
#endif

// ---------------------------------------------------------------------------
// Batched bf16 GEMM on WMMA:  C[M][N] = A[M][K] * Bw[N][K]^T
// Workgroup tile 256x64, 8 waves stacked along M, wave tile 32x64 (2x4 WMMA),
// K-step 32, double-buffered LDS (async / TDM / staged fallback).
// Grid: (N/64, M/256, batch)
// ---------------------------------------------------------------------------
__global__ __launch_bounds__(256) void gemm_bf16_wmma(
    const bf16* __restrict__ A, int lda, long long sA,
    const bf16* __restrict__ Bm, int ldb, long long sB,
    bf16* __restrict__ C, int ldc, long long sC, int K) {
  __shared__ __attribute__((aligned(16))) bf16 As[2][256][40];
  __shared__ __attribute__((aligned(16))) bf16 Bs[2][64][40];

  const int tid  = threadIdx.x;
  const int lane = tid & 31;
  const int wid  = tid >> 5;           // 8 waves along M

  A  += (long long)blockIdx.z * sA + (long long)blockIdx.y * 256 * lda;
  Bm += (long long)blockIdx.z * sB + (long long)blockIdx.x * 64 * ldb;
  C  += (long long)blockIdx.z * sC + (long long)blockIdx.y * 256 * ldc + blockIdx.x * 64;

  v8f acc[2][4] = {};

  auto loadTiles = [&](int k0, int buf) {
#if LOAD_MODE == 2
    if (wid == 0) {
      tdm_load_tile(A + k0,  lda, 256, (unsigned)(__SIZE_TYPE__)&As[buf][0][0]);
      tdm_load_tile(Bm + k0, ldb,  64, (unsigned)(__SIZE_TYPE__)&Bs[buf][0][0]);
    }
#elif LOAD_MODE == 1
#pragma unroll
    for (int i = 0; i < 4; ++i) {
      const int idx = tid + i * 256;
      const int r = idx >> 2, c = (idx & 3) << 3;
      __builtin_amdgcn_global_load_async_to_lds_b128(
          AS_GLOBAL(A + (long long)r * lda + k0 + c), AS_LDS(&As[buf][r][c]), 0, 0);
    }
    {
      const int r = tid >> 2, c = (tid & 3) << 3;
      __builtin_amdgcn_global_load_async_to_lds_b128(
          AS_GLOBAL(Bm + (long long)r * ldb + k0 + c), AS_LDS(&Bs[buf][r][c]), 0, 0);
    }
#else
    uint4 ta[4], tb;
#pragma unroll
    for (int i = 0; i < 4; ++i) {
      const int idx = tid + i * 256;
      const int r = idx >> 2, c = (idx & 3) << 3;
      ta[i] = *(const uint4*)(A + (long long)r * lda + k0 + c);
    }
    {
      const int r = tid >> 2, c = (tid & 3) << 3;
      tb = *(const uint4*)(Bm + (long long)r * ldb + k0 + c);
    }
#pragma unroll
    for (int i = 0; i < 4; ++i) {
      const int idx = tid + i * 256;
      const int r = idx >> 2, c = (idx & 3) << 3;
      *(uint4*)(&As[buf][r][c]) = ta[i];
    }
    {
      const int r = tid >> 2, c = (tid & 3) << 3;
      *(uint4*)(&Bs[buf][r][c]) = tb;
    }
#endif
  };
  auto fragA = [&](int buf, int rb) -> v16bf {
    union { uint4 q[2]; v16bf v; } u;
    const int r = rb + (lane & 15), kh = (lane >> 4) << 4;
    u.q[0] = *(const uint4*)(&As[buf][r][kh]);
    u.q[1] = *(const uint4*)(&As[buf][r][kh + 8]);
    return u.v;
  };
  auto fragB = [&](int buf, int cb) -> v16bf {
    union { uint4 q[2]; v16bf v; } u;
    const int r = cb + (lane & 15), kh = (lane >> 4) << 4;
    u.q[0] = *(const uint4*)(&Bs[buf][r][kh]);
    u.q[1] = *(const uint4*)(&Bs[buf][r][kh + 8]);
    return u.v;
  };

  const int nk = K >> 5;
  loadTiles(0, 0);
  for (int kt = 0; kt < nk; ++kt) {
#if LOAD_MODE == 1
    __builtin_amdgcn_s_wait_asynccnt(0);
#elif LOAD_MODE == 2
    if (wid == 0) __builtin_amdgcn_s_wait_tensorcnt(0);
#endif
    __syncthreads();
    const int buf = kt & 1;
    if (kt + 1 < nk) loadTiles((kt + 1) << 5, buf ^ 1);

    const v16bf a0 = fragA(buf, wid * 32);
    const v16bf a1 = fragA(buf, wid * 32 + 16);
    v16bf bfr[4];
#pragma unroll
    for (int j = 0; j < 4; ++j) bfr[j] = fragB(buf, j * 16);
#pragma unroll
    for (int j = 0; j < 4; ++j) {
      acc[0][j] = __builtin_amdgcn_wmma_f32_16x16x32_bf16(false, a0, false, bfr[j], (short)0, acc[0][j], false, false);
      acc[1][j] = __builtin_amdgcn_wmma_f32_16x16x32_bf16(false, a1, false, bfr[j], (short)0, acc[1][j], false, false);
    }
  }

  // C layout: VGPR r -> M = r + 8*(lane>=16), N = lane&15
  const int roff = (lane >> 4) << 3;
  const int col  = lane & 15;
#pragma unroll
  for (int i = 0; i < 2; ++i)
#pragma unroll
    for (int j = 0; j < 4; ++j) {
      bf16* p = C + (long long)(wid * 32 + i * 16 + roff) * ldc + j * 16 + col;
#pragma unroll
      for (int r = 0; r < 8; ++r) p[(long long)r * ldc] = (bf16)acc[i][j][r];
    }
}

// ---------------------------------------------------------------------------
// Row transform: l2norm over 512, then elu(x)+1, optionally * softplus(scale)
// ---------------------------------------------------------------------------
__global__ __launch_bounds__(256) void rowtransform(
    const bf16* __restrict__ in, bf16* __restrict__ outp,
    const float* __restrict__ scale_p, int applyScale) {
  __shared__ float sm[16];
  const long long row = blockIdx.x;
  const bf16* x = in + row * HH;
  const float v0 = (float)x[threadIdx.x];
  const float v1 = (float)x[threadIdx.x + 256];
  const float ss = blockReduceSum(v0 * v0 + v1 * v1, sm);
  const float inv = 1.f / fmaxf(sqrtf(ss), 1e-12f);
  float s = 1.f;
  if (applyScale) {
    const float sc = scale_p[0];
    s = (sc > 20.f) ? sc : log1pf(__expf(sc));   // softplus
  }
  float q0 = v0 * inv, q1 = v1 * inv;
  q0 = (q0 > 0.f) ? q0 + 1.f : __expf(q0);       // elu + 1
  q1 = (q1 > 0.f) ? q1 + 1.f : __expf(q1);
  bf16* o = outp + row * HH;
  o[threadIdx.x]       = (bf16)(q0 * s);
  o[threadIdx.x + 256] = (bf16)(q1 * s);
}

// ---------------------------------------------------------------------------
// Ksum: column sum of kp[b][2048][512] -> ksum[b][512]
// ---------------------------------------------------------------------------
__global__ __launch_bounds__(256) void colsum_bf16(
    const bf16* __restrict__ kp, float* __restrict__ ks) {
  const int b = blockIdx.y;
  const int h = blockIdx.x * 256 + threadIdx.x;
  const bf16* p = kp + (long long)b * SS * HH + h;
  float acc = 0.f;
  for (int n = 0; n < SS; ++n) {
    __builtin_prefetch(p + (long long)(n + 32) * HH, 0, 1);
    acc += (float)p[(long long)n * HH];
  }
  ks[(long long)b * HH + h] = acc;
}

// ---------------------------------------------------------------------------
// Batched transpose bf16 [R][Cc] -> bf16 [Cc][R].  Grid (Cc/32, R/32, batch),
// block (32, 8)
// ---------------------------------------------------------------------------
__global__ __launch_bounds__(256) void transpose_bf16(
    const bf16* __restrict__ in, long long sIn,
    bf16* __restrict__ outp, long long sOut, int R, int Cc) {
  __shared__ float tile[32][33];
  in   += (long long)blockIdx.z * sIn;
  outp += (long long)blockIdx.z * sOut;
  const int c0 = blockIdx.x * 32, r0 = blockIdx.y * 32;
  const int tx = threadIdx.x, ty = threadIdx.y;
#pragma unroll
  for (int i = 0; i < 32; i += 8)
    tile[ty + i][tx] = (float)in[(long long)(r0 + ty + i) * Cc + c0 + tx];
  __syncthreads();
#pragma unroll
  for (int i = 0; i < 32; i += 8)
    outp[(long long)(c0 + ty + i) * R + r0 + tx] = (bf16)tile[tx][ty + i];
}

// ---------------------------------------------------------------------------
// denom[row] = dot(qp[row][0..511], ksum[b][0..511])
// ---------------------------------------------------------------------------
__global__ __launch_bounds__(256) void denom_kernel(
    const bf16* __restrict__ qp, const float* __restrict__ ks,
    float* __restrict__ dn) {
  __shared__ float sm[16];
  const long long row = blockIdx.x;
  const int b = (int)(row >> 11);
  const bf16* q = qp + row * HH;
  const float* k = ks + (long long)b * HH;
  float a = (float)q[threadIdx.x] * k[threadIdx.x] +
            (float)q[threadIdx.x + 256] * k[threadIdx.x + 256];
  a = blockReduceSum(a, sm);
  if (threadIdx.x == 0) dn[row] = a;
}

// ---------------------------------------------------------------------------
// Epilogue: out = 0.1*outf/denom + tok, then LayerNorm2 -> tok (in place)
// ---------------------------------------------------------------------------
__global__ __launch_bounds__(256) void epilogue_ln2(
    const bf16* __restrict__ outf, const float* __restrict__ dnm,
    const float* __restrict__ w, const float* __restrict__ bb,
    float* __restrict__ tok) {
  __shared__ float sm[16];
  const long long row = blockIdx.x;
  const float dscale = 0.1f / (dnm[row] + ATTN_EPS);
  const bf16* of = outf + row * NP;
  float* tk = tok + row * KP;
  float v[3], s = 0.f, ss = 0.f;
#pragma unroll
  for (int i = 0; i < 3; ++i) {
    const int t = threadIdx.x + i * 256;
    float val = 0.f;
    if (t < TT) val = (float)of[t] * dscale + tk[t];
    v[i] = val; s += val; ss += val * val;
  }
  s = blockReduceSum(s, sm);
  ss = blockReduceSum(ss, sm);
  const float mean = s * (1.f / TT);
  const float var  = ss * (1.f / TT) - mean * mean;
  const float inv  = rsqrtf(var + LN_EPS);
#pragma unroll
  for (int i = 0; i < 3; ++i) {
    const int t = threadIdx.x + i * 256;
    if (t < KP) tk[t] = (t < TT) ? (v[i] - mean) * inv * w[t] + bb[t] : 0.f;
  }
}

// ---------------------------------------------------------------------------
// Final prediction: out[b] = dot(tok[b][2047][0..512], pred_w)
// ---------------------------------------------------------------------------
__global__ __launch_bounds__(256) void predict_kernel(
    const float* __restrict__ tok, const float* __restrict__ pw,
    float* __restrict__ outp) {
  __shared__ float sm[16];
  const int b = blockIdx.x;
  const float* x = tok + ((long long)b * SS + (SS - 1)) * KP;
  float a = 0.f;
#pragma unroll
  for (int i = 0; i < 3; ++i) {
    const int t = threadIdx.x + i * 256;
    if (t < TT) a += x[t] * pw[t];
  }
  a = blockReduceSum(a, sm);
  if (threadIdx.x == 0) outp[b] = a;
}

// ---------------------------------------------------------------------------
// Host launcher
// ---------------------------------------------------------------------------
extern "C" void kernel_launch(void* const* d_in, const int* in_sizes, int n_in,
                              void* d_out, int out_size, void* d_ws, size_t ws_size,
                              hipStream_t stream) {
  (void)in_sizes; (void)n_in; (void)out_size; (void)ws_size;

  const float* xs    = (const float*)d_in[0];
  const float* ys    = (const float*)d_in[1];
  const float* qx    = (const float*)d_in[2];
  const float* Wq    = (const float*)d_in[3];
  const float* Wk    = (const float*)d_in[4];
  const float* Wv    = (const float*)d_in[5];
  const float* Wo    = (const float*)d_in[6];
  const float* ln1w  = (const float*)d_in[7];
  const float* ln1b  = (const float*)d_in[8];
  const float* ln2w  = (const float*)d_in[9];
  const float* ln2b  = (const float*)d_in[10];
  const float* scale = (const float*)d_in[11];
  const float* predw = (const float*)d_in[12];
  float* out = (float*)d_out;

  char* base = (char*)d_ws;
  size_t off = 0;
  auto alloc = [&](size_t bytes) -> char* {
    char* p = base + off;
    off += (bytes + 255) & ~(size_t)255;
    return p;
  };

  float* tok = (float*)alloc((size_t)BB * SS * KP * 4);
  bf16*  xln = (bf16*)alloc((size_t)BB * SS * KP * 2);
  bf16*  qf  = (bf16*)alloc((size_t)BB * SS * HH * 2);
  bf16*  kf  = (bf16*)alloc((size_t)BB * SS * HH * 2);
  bf16*  vf  = (bf16*)alloc((size_t)BB * SS * NP * 2);
  bf16*  qp  = (bf16*)alloc((size_t)BB * SS * HH * 2);
  bf16*  kp  = (bf16*)alloc((size_t)BB * SS * HH * 2);
  bf16*  kpt = (bf16*)alloc((size_t)BB * HH * SS * 2);
  bf16*  vt  = (bf16*)alloc((size_t)BB * NP * SS * 2);
  float* ksum  = (float*)alloc((size_t)BB * HH * 4);
  float* denom = (float*)alloc((size_t)BB * SS * 4);
  bf16*  wqb = (bf16*)alloc((size_t)HH * KP * 2);
  bf16*  wkb = (bf16*)alloc((size_t)HH * KP * 2);
  bf16*  wvb = (bf16*)alloc((size_t)NP * KP * 2);
  bf16*  wob = (bf16*)alloc((size_t)NP * NP * 2);
  // Aliased scratch (regions provably dead at time of reuse):
  bf16* kvb  = (bf16*)kf;                         // KV  [b][512][576] bf16
  bf16* w2b  = (bf16*)qf;                         // W2  [b][512][576] bf16
  bf16* w2t  = (bf16*)((char*)kp + 67108864);     // W2t [b][576][512] bf16
  bf16* outf = (bf16*)vf;                         // out [b][2048][576] bf16

  const int ROWS = BB * SS;

  build_tokens<<<ROWS, 256, 0, stream>>>(xs, ys, qx, tok);

  for (int l = 0; l < LL; ++l) {
    const float* wq = Wq + (size_t)l * HH * TT;
    const float* wk = Wk + (size_t)l * HH * TT;
    const float* wv = Wv + (size_t)l * TT * TT;
    const float* wo = Wo + (size_t)l * TT * TT;

    // weight pad/convert to bf16
    padconv<<<(HH * KP + 255) / 256, 256, 0, stream>>>(wq, HH, TT, wqb, HH, KP);
    padconv<<<(HH * KP + 255) / 256, 256, 0, stream>>>(wk, HH, TT, wkb, HH, KP);
    padconv<<<(NP * KP + 255) / 256, 256, 0, stream>>>(wv, TT, TT, wvb, NP, KP);
    padconv<<<(NP * NP + 255) / 256, 256, 0, stream>>>(wo, TT, TT, wob, NP, NP);

    // LN1 -> x (bf16, padded)
    ln_to_bf16<<<ROWS, 256, 0, stream>>>(tok, ln1w + (size_t)l * TT,
                                         ln1b + (size_t)l * TT, xln);

    // Q = x*Wq^T, K = x*Wk^T, V = x*Wv^T   (M=2048 per batch)
    gemm_bf16_wmma<<<dim3(HH / 64, SS / 256, BB), 256, 0, stream>>>(
        xln, KP, (long long)SS * KP, wqb, KP, 0, qf, HH, (long long)SS * HH, KP);
    gemm_bf16_wmma<<<dim3(HH / 64, SS / 256, BB), 256, 0, stream>>>(
        xln, KP, (long long)SS * KP, wkb, KP, 0, kf, HH, (long long)SS * HH, KP);
    gemm_bf16_wmma<<<dim3(NP / 64, SS / 256, BB), 256, 0, stream>>>(
        xln, KP, (long long)SS * KP, wvb, KP, 0, vf, NP, (long long)SS * NP, KP);

    // feature maps
    rowtransform<<<ROWS, 256, 0, stream>>>(qf, qp, scale + l, 1);
    rowtransform<<<ROWS, 256, 0, stream>>>(kf, kp, scale + l, 0);

    // Ksum, then transposes for the reduction GEMMs
    colsum_bf16<<<dim3(HH / 256, BB), 256, 0, stream>>>(kp, ksum);
    transpose_bf16<<<dim3(HH / 32, SS / 32, BB), dim3(32, 8), 0, stream>>>(
        kp, (long long)SS * HH, kpt, (long long)HH * SS, SS, HH);
    transpose_bf16<<<dim3(NP / 32, SS / 32, BB), dim3(32, 8), 0, stream>>>(
        vf, (long long)SS * NP, vt, (long long)NP * SS, SS, NP);

    // KV[h][u] = sum_n Kp[n][h] * V[n][u]   (M=512, N=576, K=2048)
    gemm_bf16_wmma<<<dim3(NP / 64, HH / 256, BB), 256, 0, stream>>>(
        kpt, SS, (long long)HH * SS, vt, SS, (long long)NP * SS,
        kvb, NP, (long long)HH * NP, SS);

    // W2[h][t] = sum_u KV[h][u] * Wo[t][u]  (M=512, N=576, K=576)
    gemm_bf16_wmma<<<dim3(NP / 64, HH / 256, BB), 256, 0, stream>>>(
        kvb, NP, (long long)HH * NP, wob, NP, 0,
        w2b, NP, (long long)HH * NP, NP);
    transpose_bf16<<<dim3(NP / 32, HH / 32, BB), dim3(32, 8), 0, stream>>>(
        w2b, (long long)HH * NP, w2t, (long long)NP * HH, HH, NP);

    // denom[n] = Qp[n] . Ksum
    denom_kernel<<<ROWS, 256, 0, stream>>>(qp, ksum, denom);

    // outf[n][t] = sum_h Qp[n][h] * W2[h][t]  (M=2048, N=576, K=512)
    gemm_bf16_wmma<<<dim3(NP / 64, SS / 256, BB), 256, 0, stream>>>(
        qp, HH, (long long)SS * HH, w2t, HH, (long long)NP * HH,
        outf, NP, (long long)SS * NP, HH);

    // residual + LN2 -> tok
    epilogue_ln2<<<ROWS, 256, 0, stream>>>(outf, denom, ln2w + (size_t)l * TT,
                                           ln2b + (size_t)l * TT, tok);
  }

  predict_kernel<<<BB, 256, 0, stream>>>(tok, predw, out);
}
